// GNNRecommender_3092376453606
// MI455X (gfx1250) — matrix operand order repaired and checked by reading, not compiled, not run on hardware
//
#include <hip/hip_runtime.h>

typedef __bf16 bf16;
typedef __attribute__((ext_vector_type(16))) __bf16 v16bf;
typedef __attribute__((ext_vector_type(8)))  __bf16 v8bf;
typedef __attribute__((ext_vector_type(8)))  float  v8f;

#define NEG_SLOPE 0.2f
#define LN_EPS 1e-5f

// ---------------- float <-> monotone uint encoding for atomicMax ----------------
__device__ __forceinline__ unsigned encf(float f) {
  unsigned u = __float_as_uint(f);
  return (u & 0x80000000u) ? ~u : (u | 0x80000000u);
}
__device__ __forceinline__ float decf(unsigned u) {
  u = (u & 0x80000000u) ? (u & 0x7fffffffu) : ~u;
  return __uint_as_float(u);
}

// ---------------- fills ----------------
__global__ void k_fillf(float* p, float v, long n) {
  long i = (long)blockIdx.x * blockDim.x + threadIdx.x;
  if (i < n) p[i] = v;
}
__global__ void k_fillu(unsigned* p, unsigned v, long n) {
  long i = (long)blockIdx.x * blockDim.x + threadIdx.x;
  if (i < n) p[i] = v;
}

// ---------------- fp32 -> bf16 convert (A operand), 4 elements/thread ----------
__global__ void k_cvt_bf16(const float* __restrict__ src, bf16* __restrict__ dst, long n) {
  long i = ((long)blockIdx.x * blockDim.x + threadIdx.x) * 4;
  if (i + 3 < n) {
    const float4 f = *(const float4*)(src + i);
    dst[i]     = (bf16)f.x;
    dst[i + 1] = (bf16)f.y;
    dst[i + 2] = (bf16)f.z;
    dst[i + 3] = (bf16)f.w;
  } else {
    for (; i < n; ++i) dst[i] = (bf16)src[i];
  }
}

// ---------------- W[K,Nc] fp32 -> Wt[Nc,K] bf16 (B operand, K-contiguous) -------
__global__ void k_transpose_w_bf16(const float* __restrict__ W, bf16* __restrict__ Wt,
                                   int K, int Nc) {
  const int idx = blockIdx.x * blockDim.x + threadIdx.x;
  if (idx >= K * Nc) return;
  const int k = idx / Nc, c = idx - k * Nc;
  Wt[(size_t)c * K + k] = (bf16)W[idx];
}

// ---------------- WMMA bf16 GEMM: C[Mpad,Nc] = A[M,K] @ W[K,Nc], fp32 accumulate --
// K, Nc compile-time: k-loop fully unrolls, address math is shifts + immediates.
// A bf16 row-major (loads clamped to row M-1); C padded to Mpad = 32*mtiles rows,
// so stores are unconditional. One wave = 32x64 tile, 2 A frags x 4 B frags =
// 8 WMMAs per k-step; all operand loads are b128.
template <int K, int Nc>
__global__ void k_gemm_wmma_bf16(const bf16* __restrict__ A, const bf16* __restrict__ Wt,
                                 float* __restrict__ C, int M) {
  const int lane = threadIdx.x & 31;
  const int wave = threadIdx.x >> 5;
  const int mtiles  = (M + 31) >> 5;           // 32 rows per wave
  constexpr int nstrips = Nc >> 6;             // 64 cols per wave
  const int wg = blockIdx.x * (blockDim.x >> 5) + wave;
  if (wg >= mtiles * nstrips) return;
  const int mt = wg % mtiles;
  const int ns = wg / mtiles;
  const int mbase = mt << 5;
  const int nbase = ns << 6;
  const int half = lane >> 4;                  // 0: lanes 0-15, 1: lanes 16-31
  const int lrow = lane & 15;

  v8f acc[2][4];
#pragma unroll
  for (int m = 0; m < 2; ++m)
#pragma unroll
    for (int t = 0; t < 4; ++t) acc[m][t] = (v8f){0, 0, 0, 0, 0, 0, 0, 0};

  // A rows for this lane (clamped; padded output rows are never consumed)
  const int r0 = mbase + lrow;
  const int r1 = mbase + 16 + lrow;
  const bf16* A0 = A + (size_t)(r0 < M ? r0 : M - 1) * K;
  const bf16* A1 = A + (size_t)(r1 < M ? r1 : M - 1) * K;

  const int base_k = half * 16;                // A: two 8-elem runs at base_k, base_k+8
  const int kgrp   = half * 16;                // B: one 16-elem run at kgrp
  const bf16* Wl = Wt + (size_t)(nbase + lrow) * K + kgrp;

#pragma unroll
  for (int k0 = 0; k0 < K; k0 += 32) {
    v16bf a[2];
#pragma unroll
    for (int m = 0; m < 2; ++m) {
      const bf16* Ar = (m == 0) ? A0 : A1;
      const v8bf lo = *(const v8bf*)(Ar + k0 + base_k);        // K = base_k .. +7
      const v8bf hi = *(const v8bf*)(Ar + k0 + base_k + 8);    // K = base_k+8 .. +15
#pragma unroll
      for (int j = 0; j < 8; ++j) { a[m][j] = lo[j]; a[m][8 + j] = hi[j]; }
    }
#pragma unroll
    for (int t = 0; t < 4; ++t) {
      const v16bf b = *(const v16bf*)(Wl + (size_t)t * 16 * K + k0);
      acc[0][t] = __builtin_amdgcn_wmma_f32_16x16x32_bf16(false, a[0], false, b, (short)0, acc[0][t], false, false);
      acc[1][t] = __builtin_amdgcn_wmma_f32_16x16x32_bf16(false, a[1], false, b, (short)0, acc[1][t], false, false);
    }
  }

  // store: unconditional (C has Mpad rows); lane n = lrow, VGPR r -> +r rows
#pragma unroll
  for (int m = 0; m < 2; ++m) {
#pragma unroll
    for (int t = 0; t < 4; ++t) {
      float* Cp = C + (size_t)(mbase + m * 16 + half * 8) * Nc + (nbase + t * 16 + lrow);
#pragma unroll
      for (int r = 0; r < 8; ++r) Cp[(size_t)r * Nc] = acc[m][t][r];
    }
  }
}

// ---------------- attention logits: a_src/a_dst [n,H] = dot(xp[n,h,:], att[h,:]) ----
template <int H, int CC>
__global__ void k_attn_logits(const float* __restrict__ xp,
                              const float* __restrict__ att_s, const float* __restrict__ att_d,
                              float* __restrict__ asrc, float* __restrict__ adst, int n) {
  const int i = blockIdx.x * blockDim.x + threadIdx.x;
  if (i >= n * H) return;
  const int node = i / H, h = i % H;
  const float* xr = xp + (size_t)node * H * CC + (size_t)h * CC;
  float s0 = 0.f, s1 = 0.f;
#pragma unroll 4
  for (int c = 0; c < CC; ++c) {
    const float v = xr[c];
    s0 += v * att_s[h * CC + c];
    s1 += v * att_d[h * CC + c];
  }
  asrc[i] = s0;
  adst[i] = s1;
}

__device__ __forceinline__ void edge_nodes(const int* __restrict__ ei, int E, int e, int& s, int& d) {
  if (e < E) { s = ei[e]; d = ei[E + e]; }
  else       { s = d = e - E; }                 // implicit self-loops
}

template <int H>
__device__ __forceinline__ float edge_logit(const int* __restrict__ ei, int E,
                                            const float* __restrict__ asrc,
                                            const float* __restrict__ adst,
                                            int e, int h, int& d) {
  int s;
  edge_nodes(ei, E, e, s, d);
  float lg = asrc[(size_t)s * H + h] + adst[(size_t)d * H + h];
  return lg > 0.f ? lg : NEG_SLOPE * lg;
}

// ---------------- edge pass 1: segment max of leaky_relu logits ----------------
template <int H>
__global__ void k_edge_max(const int* __restrict__ ei, int E, int n,
                           const float* __restrict__ asrc, const float* __restrict__ adst,
                           unsigned* __restrict__ menc) {
  const long idx = (long)blockIdx.x * blockDim.x + threadIdx.x;
  if (idx >= (long)(E + n) * H) return;
  const int e = (int)(idx / H), h = (int)(idx - (long)e * H);
  int d;
  const float lg = edge_logit<H>(ei, E, asrc, adst, e, h, d);
  atomicMax(&menc[(size_t)d * H + h], encf(lg));
}

// ---------------- edge pass 2: segment sum of exp(e - m) ----------------
template <int H>
__global__ void k_edge_expsum(const int* __restrict__ ei, int E, int n,
                              const float* __restrict__ asrc, const float* __restrict__ adst,
                              const unsigned* __restrict__ menc, float* __restrict__ ssum) {
  const long idx = (long)blockIdx.x * blockDim.x + threadIdx.x;
  if (idx >= (long)(E + n) * H) return;
  const int e = (int)(idx / H), h = (int)(idx - (long)e * H);
  int d;
  const float lg = edge_logit<H>(ei, E, asrc, adst, e, h, d);
  const float mm = decf(menc[(size_t)d * H + h]);
  atomicAdd(&ssum[(size_t)d * H + h], __expf(lg - mm));
}

// ---------------- edge pass 3: alpha = exp(e - m) / s[dst] ----------------
template <int H>
__global__ void k_edge_alpha(const int* __restrict__ ei, int E, int n,
                             const float* __restrict__ asrc, const float* __restrict__ adst,
                             const unsigned* __restrict__ menc, const float* __restrict__ ssum,
                             float* __restrict__ alpha) {
  const long idx = (long)blockIdx.x * blockDim.x + threadIdx.x;
  if (idx >= (long)(E + n) * H) return;
  const int e = (int)(idx / H), h = (int)(idx - (long)e * H);
  int d;
  const float lg = edge_logit<H>(ei, E, asrc, adst, e, h, d);
  const float mm = decf(menc[(size_t)d * H + h]);
  alpha[idx] = __expf(lg - mm) / ssum[(size_t)d * H + h];
}

// ---------------- edge pass 4 (heavy): agg[dst] += alpha * xp[src] ----------------
// All operands L2-resident (<= ~65MB working set vs 192MB L2); f32 atomics hit L2.
template <int H, int CC>
__global__ void k_edge_agg(const int* __restrict__ ei, int E, int n,
                           const float* __restrict__ xp, const float* __restrict__ alpha,
                           float* __restrict__ agg) {
  constexpr int HC = H * CC;
  const long gid = (long)blockIdx.x * blockDim.x + threadIdx.x;
  if (gid >= (long)(E + n) * HC) return;
  const int e = (int)(gid / HC);
  const int c = (int)(gid - (long)e * HC);
  const int h = c / CC;
  int s, d;
  edge_nodes(ei, E, e, s, d);
  const float a = alpha[(long)e * H + h];
  atomicAdd(&agg[(size_t)d * HC + c], a * xp[(size_t)s * HC + c]);
}

// ---------------- bias + LayerNorm + ELU, one wave32 per node ----------------
template <int D>
__global__ void k_node_ln_elu(const float* __restrict__ agg, const float* __restrict__ bias,
                              const float* __restrict__ g, const float* __restrict__ b,
                              float* __restrict__ out, int n) {
  const int wave = threadIdx.x >> 5;
  const int lane = threadIdx.x & 31;
  const int node = blockIdx.x * (blockDim.x >> 5) + wave;
  if (node >= n) return;
  const float* row = agg + (size_t)node * D;
  constexpr int PER = D >> 5;                   // 8 (D=256) or 4 (D=128)
  float vals[PER];
  float s0 = 0.f, s1 = 0.f;
#pragma unroll
  for (int j = 0; j < PER; ++j) {
    const int c = lane + j * 32;
    const float v = row[c] + bias[c];
    vals[j] = v;
    s0 += v; s1 += v * v;
  }
#pragma unroll
  for (int o = 16; o > 0; o >>= 1) {
    s0 += __shfl_xor(s0, o, 32);
    s1 += __shfl_xor(s1, o, 32);
  }
  const float mu = s0 / D;
  const float var = s1 / D - mu * mu;
  const float rstd = rsqrtf(var + LN_EPS);
  float* orow = out + (size_t)node * D;
#pragma unroll
  for (int j = 0; j < PER; ++j) {
    const int c = lane + j * 32;
    const float v = (vals[j] - mu) * rstd * g[c] + b[c];
    orow[c] = v > 0.f ? v : (__expf(v) - 1.f);  // ELU(alpha=1)
  }
}

// ---------------- final layer: out = agg + bias ----------------
template <int D>
__global__ void k_add_bias(const float* __restrict__ agg, const float* __restrict__ bias,
                           float* __restrict__ out, long n) {
  const long i = (long)blockIdx.x * blockDim.x + threadIdx.x;
  if (i >= n * D) return;
  out[i] = agg[i] + bias[(int)(i & (D - 1))];
}

// =========================================================================

static inline unsigned blocks_for(long n, int tb) { return (unsigned)((n + tb - 1) / tb); }

struct LayerBufs {
  const int* ei;
  int N, E, E2;
  float *XP, *AGG, *ASRC, *ADST, *SSUM, *ALPHA;
  unsigned* MENC;
  bf16 *ABF, *WT;
};

template <int K, int H, int CC, bool DO_LN>
static void run_layer(const LayerBufs& L, const float* in, const float* W,
                      const float* as, const float* ad, const float* bias,
                      const float* lg, const float* lb, float* outbuf, hipStream_t stream) {
  constexpr int HC = H * CC;
  const int TB = 256;
  const int N = L.N, E = L.E, E2 = L.E2;
  // operand prep: A fp32 -> bf16 row-major; W fp32 [K,HC] -> bf16 Wt [HC,K]
  k_cvt_bf16<<<blocks_for(((long)N * K + 3) / 4, TB), TB, 0, stream>>>(in, L.ABF, (long)N * K);
  k_transpose_w_bf16<<<blocks_for((long)K * HC, TB), TB, 0, stream>>>(W, L.WT, K, HC);
  // GEMM via WMMA bf16 (32x64 tile per wave, 8 wmma per k-step, K-loop unrolled)
  const int mtiles = (N + 31) / 32;
  const long waves = (long)mtiles * (HC / 64);
  k_gemm_wmma_bf16<K, HC><<<blocks_for(waves, 8), TB, 0, stream>>>(L.ABF, L.WT, L.XP, N);
  // init softmax state + accumulator
  k_fillu<<<blocks_for((long)N * H, TB), TB, 0, stream>>>(L.MENC, 0u, (long)N * H);
  k_fillf<<<blocks_for((long)N * H, TB), TB, 0, stream>>>(L.SSUM, 0.f, (long)N * H);
  k_fillf<<<blocks_for((long)N * HC, TB), TB, 0, stream>>>(L.AGG, 0.f, (long)N * HC);
  // logits, edge softmax, aggregation
  k_attn_logits<H, CC><<<blocks_for((long)N * H, TB), TB, 0, stream>>>(L.XP, as, ad, L.ASRC, L.ADST, N);
  k_edge_max<H>   <<<blocks_for((long)E2 * H, TB), TB, 0, stream>>>(L.ei, E, N, L.ASRC, L.ADST, L.MENC);
  k_edge_expsum<H><<<blocks_for((long)E2 * H, TB), TB, 0, stream>>>(L.ei, E, N, L.ASRC, L.ADST, L.MENC, L.SSUM);
  k_edge_alpha<H> <<<blocks_for((long)E2 * H, TB), TB, 0, stream>>>(L.ei, E, N, L.ASRC, L.ADST, L.MENC, L.SSUM, L.ALPHA);
  k_edge_agg<H, CC><<<blocks_for((long)E2 * HC, TB), TB, 0, stream>>>(L.ei, E, N, L.XP, L.ALPHA, L.AGG);
  // epilogue
  if (DO_LN)
    k_node_ln_elu<HC><<<blocks_for((long)N, 8), TB, 0, stream>>>(L.AGG, bias, lg, lb, outbuf, N);
  else
    k_add_bias<HC><<<blocks_for((long)N * HC, TB), TB, 0, stream>>>(L.AGG, bias, outbuf, (long)N);
}

extern "C" void kernel_launch(void* const* d_in, const int* in_sizes, int n_in,
                              void* d_out, int out_size, void* d_ws, size_t ws_size,
                              hipStream_t stream) {
  const float* x   = (const float*)d_in[0];
  const int*   ei  = (const int*)d_in[1];
  const float* W1  = (const float*)d_in[2];
  const float* as1 = (const float*)d_in[3];
  const float* ad1 = (const float*)d_in[4];
  const float* b1  = (const float*)d_in[5];
  const float* g1  = (const float*)d_in[6];
  const float* lb1 = (const float*)d_in[7];
  const float* W2  = (const float*)d_in[8];
  const float* as2 = (const float*)d_in[9];
  const float* ad2 = (const float*)d_in[10];
  const float* b2  = (const float*)d_in[11];
  const float* g2  = (const float*)d_in[12];
  const float* lb2 = (const float*)d_in[13];
  const float* W3  = (const float*)d_in[14];
  const float* as3 = (const float*)d_in[15];
  const float* ad3 = (const float*)d_in[16];
  const float* b3  = (const float*)d_in[17];

  const int N = in_sizes[0] / 128;
  const int E = in_sizes[1] / 2;
  const int E2 = E + N;
  const int Npad = ((N + 31) / 32) * 32;        // GEMM output padded to tile rows

  float* ws = (float*)d_ws;
  size_t off = 0;
  float*    XP    = ws + off; off += (size_t)Npad * 256;
  float*    AGG   = ws + off; off += (size_t)N * 256;
  float*    HB    = ws + off; off += (size_t)N * 256;
  float*    ASRC  = ws + off; off += (size_t)N * 4;
  float*    ADST  = ws + off; off += (size_t)N * 4;
  unsigned* MENC  = (unsigned*)(ws + off); off += (size_t)N * 4;
  float*    SSUM  = ws + off; off += (size_t)N * 4;
  float*    ALPHA = ws + off; off += (size_t)E2 * 4;
  bf16*     ABF   = (bf16*)(ws + off); off += (size_t)N * 128;   // N*256 bf16
  bf16*     WT    = (bf16*)(ws + off); off += 16384;             // 32768 bf16

  LayerBufs L{ei, N, E, E2, XP, AGG, ASRC, ADST, SSUM, ALPHA, MENC, ABF, WT};

  // Layer 1: [N,128] -> 4 heads x 64, concat -> [N,256], LN+ELU
  run_layer<128, 4, 64, true>(L, x, W1, as1, ad1, b1, g1, lb1, HB, stream);
  // Layer 2: [N,256] -> 2 heads x 64, concat -> [N,128], LN+ELU
  run_layer<256, 2, 64, true>(L, HB, W2, as2, ad2, b2, g2, lb2, HB, stream);
  // Layer 3: [N,128] -> 1 head x 128 -> [N,128] + bias
  run_layer<128, 1, 128, false>(L, HB, W3, as3, ad3, b3, nullptr, nullptr, (float*)d_out, stream);
}
